// DecoderRNNWithAttention_73675868996266
// MI455X (gfx1250) — matrix operand order repaired and checked by reading, not compiled, not run on hardware
//
#include <hip/hip_runtime.h>
#include <hip/hip_bf16.h>

// ---------------------------------------------------------------------------
// Problem constants
// ---------------------------------------------------------------------------
#define BB   64
#define PP   196
#define ED_  2048
#define AD_  512
#define DD_  512
#define EE   300
#define VV   1000
#define TT   96
#define TDEC 95

typedef __attribute__((ext_vector_type(16))) __bf16 v16bf;
typedef __attribute__((ext_vector_type(8)))  float  v8f;

__device__ __forceinline__ float sigmoidf_(float x) { return 1.0f / (1.0f + __expf(-x)); }

// ---------------------------------------------------------------------------
// 1) stable argsort(-lens) + caption gather + scalar outputs
// ---------------------------------------------------------------------------
__global__ void k_sort(const int* __restrict__ lens, const int* __restrict__ caps,
                       int* __restrict__ sortind, int* __restrict__ declen,
                       int* __restrict__ caps_s,
                       float* __restrict__ out_caps, float* __restrict__ out_declen,
                       float* __restrict__ out_sortind) {
  __shared__ int s_sort[BB];
  int tid = threadIdx.x;
  if (tid < BB) {
    int li = lens[tid];
    int rank = 0;
    for (int j = 0; j < BB; ++j) {
      int lj = lens[j];
      if (lj > li || (lj == li && j < tid)) rank++;
    }
    s_sort[rank] = tid;
  }
  __syncthreads();
  if (tid < BB) {
    int src = s_sort[tid];
    sortind[tid] = src;
    int dl = lens[src] - 1;
    declen[tid] = dl;
    out_declen[tid] = (float)dl;
    out_sortind[tid] = (float)src;
  }
  __syncthreads();
  for (int i = tid; i < BB * TT; i += blockDim.x) {
    int b = i / TT, j = i % TT;
    int v = caps[s_sort[b] * TT + j];
    caps_s[i] = v;
    out_caps[i] = (float)v;
  }
}

// ---------------------------------------------------------------------------
// 2) fp32 weight (K x N, row-major) -> bf16 in WMMA B-fragment swizzled order.
//    dst[(nt*Kt + kt)*512 + lane*16 + j] = W[k,n],
//    k = kt*32 + (lane>>4)*16 + j,  n = nt*16 + (lane&15); zero-padded.
// ---------------------------------------------------------------------------
__global__ void k_convert(const float* __restrict__ src, __bf16* __restrict__ dst,
                          int K, int N, int Kt, int ntiles) {
  int total = Kt * ntiles * 512;
  for (int idx = blockIdx.x * blockDim.x + threadIdx.x; idx < total;
       idx += gridDim.x * blockDim.x) {
    int r = idx & 511;
    int tile = idx >> 9;
    int kt = tile % Kt, nt = tile / Kt;
    int lane = r >> 4, j = r & 15;
    int k = kt * 32 + ((lane >> 4) << 4) + j;
    int n = nt * 16 + (lane & 15);
    float v = (k < K && n < N) ? src[(size_t)k * N + n] : 0.0f;
    dst[idx] = (__bf16)v;
  }
}

// ---------------------------------------------------------------------------
// 3) mean over P of sorted encoder_out
// ---------------------------------------------------------------------------
__global__ void k_mean(const float* __restrict__ eo, const int* __restrict__ sortind,
                       float* __restrict__ mean) {
  int idx = blockIdx.x * blockDim.x + threadIdx.x;
  if (idx >= BB * ED_) return;
  int b = idx >> 11;           // /2048
  int d = idx & (ED_ - 1);
  const float* row = eo + (size_t)sortind[b] * PP * ED_ + d;
  float s = 0.0f;
  for (int p = 0; p < PP; ++p) s += row[(size_t)p * ED_];
  mean[idx] = s * (1.0f / (float)PP);
}

// ---------------------------------------------------------------------------
// 4) embedding gather for sorted captions
// ---------------------------------------------------------------------------
__global__ void k_embs(const float* __restrict__ emb, const int* __restrict__ caps_s,
                       float* __restrict__ embs) {
  int idx = blockIdx.x * blockDim.x + threadIdx.x;
  if (idx >= BB * TT * EE) return;
  int bt = idx / EE;
  int e  = idx - bt * EE;
  embs[idx] = emb[(size_t)caps_s[bt] * EE + e];
}

// ---------------------------------------------------------------------------
// 5) Generic WMMA bf16 GEMM:  C = rowScale * (A(f32) @ Wsw(bf16) + bias + add)
//    A: M x lda fp32 row-major (converted to bf16 on the fly, built into the
//    contiguous-K A-fragment order k = 16*half + s, matching Wsw). Wsw pre-swizzled B fragments.
//    Optional sort indirection for A rows (phys = sortind[m/groupP]*groupP+m%groupP).
//    Optional length mask: row *= (declen[row] > t).
//    One wave computes one 16x16 M-tile x 4 N-tiles. Inner loop is branch-free:
//    out-of-range N-tiles are clamped for loads/WMMA and only skipped at store.
// ---------------------------------------------------------------------------
#define WMMA_BF16(ACC, A_, B_)                                                  \
  ACC = __builtin_amdgcn_wmma_f32_16x16x32_bf16(false, A_, false, B_, (short)0, \
                                                ACC, false, false)

__global__ void k_gemm(const float* __restrict__ A, int lda,
                       const int* __restrict__ sortind, int groupP,
                       const __bf16* __restrict__ Wsw,
                       const float* __restrict__ bias,
                       const float* __restrict__ addsrc, int ldadd,
                       float* __restrict__ C, int ldc,
                       int M, int Kt, int ntiles, int Nout,
                       const int* __restrict__ declen, int t) {
  int lane = threadIdx.x & 31;
  // Force the wave-tile coordinates into SGPRs: all control flow and tile
  // addressing below is provably uniform -> no exec-mask churn around WMMA.
  int wid = __builtin_amdgcn_readfirstlane((blockIdx.x * blockDim.x + threadIdx.x) >> 5);
  int mtiles = M >> 4;
  int ntg    = (ntiles + 3) >> 2;
  if (wid >= mtiles * ntg) return;
  int mt  = wid % mtiles;
  int nt0 = (wid / mtiles) << 2;

  // clamped tile indices (loads stay in-bounds; extra WMMA results unstored)
  int ntc0 = nt0 + 0 < ntiles ? nt0 + 0 : ntiles - 1;
  int ntc1 = nt0 + 1 < ntiles ? nt0 + 1 : ntiles - 1;
  int ntc2 = nt0 + 2 < ntiles ? nt0 + 2 : ntiles - 1;
  int ntc3 = nt0 + 3 < ntiles ? nt0 + 3 : ntiles - 1;

  int m = (mt << 4) + (lane & 15);
  int mphys = m;
  if (groupP) mphys = sortind[m / groupP] * groupP + (m % groupP);
  const float* Arow = A + (size_t)mphys * lda;
  int aoff = (lane >> 4) << 4;  // K offset for upper lane half

  v8f acc0 = {0, 0, 0, 0, 0, 0, 0, 0};
  v8f acc1 = acc0, acc2 = acc0, acc3 = acc0;

  const __bf16* Wl = Wsw + (size_t)lane * 16;
  const __bf16* p0 = Wl + (size_t)ntc0 * Kt * 512;
  const __bf16* p1 = Wl + (size_t)ntc1 * Kt * 512;
  const __bf16* p2 = Wl + (size_t)ntc2 * Kt * 512;
  const __bf16* p3 = Wl + (size_t)ntc3 * Kt * 512;

  for (int kt = 0; kt < Kt; ++kt) {
    // B fragments: 4 contiguous 32B vector loads (one clause)
    v16bf b0 = *reinterpret_cast<const v16bf*>(p0);
    v16bf b1 = *reinterpret_cast<const v16bf*>(p1);
    v16bf b2 = *reinterpret_cast<const v16bf*>(p2);
    v16bf b3 = *reinterpret_cast<const v16bf*>(p3);
    // A fragment (contiguous K order: k = 16*half + s, matching Wsw)
    v16bf a;
    int kb = (kt << 5) + aoff;
#pragma unroll
    for (int p = 0; p < 8; ++p) {
      int k0 = kb + (p << 1);
      float2 f = *reinterpret_cast<const float2*>(Arow + k0);
      a[2 * p]     = (__bf16)f.x;
      a[2 * p + 1] = (__bf16)f.y;
    }
    WMMA_BF16(acc0, a, b0);
    WMMA_BF16(acc1, a, b1);
    WMMA_BF16(acc2, a, b2);
    WMMA_BF16(acc3, a, b3);
    p0 += 512; p1 += 512; p2 += 512; p3 += 512;
  }

#define STORE_TILE(ACC, I)                                                     \
  do {                                                                         \
    int nt = nt0 + (I);                                                        \
    if (nt < ntiles) {                                                         \
      int n = (nt << 4) + (lane & 15);                                         \
      if (n < Nout) {                                                          \
        float bv = bias ? bias[n] : 0.0f;                                      \
        _Pragma("unroll") for (int r = 0; r < 8; ++r) {                        \
          int mrow = (mt << 4) + ((lane >> 4) << 3) + r;                       \
          float v = ACC[r] + bv;                                               \
          if (addsrc) v += addsrc[(size_t)mrow * ldadd + n];                   \
          if (declen) v *= (declen[mrow] > t) ? 1.0f : 0.0f;                   \
          C[(size_t)mrow * ldc + n] = v;                                       \
        }                                                                      \
      }                                                                        \
    }                                                                          \
  } while (0)

  STORE_TILE(acc0, 0);
  STORE_TILE(acc1, 1);
  STORE_TILE(acc2, 2);
  STORE_TILE(acc3, 3);
#undef STORE_TILE
}

// ---------------------------------------------------------------------------
// 6) attention: e = relu(att1 + att2) . w_full + b_full ; softmax over P;
//    alphas output (masked); awe = sum_p alpha * eo   (one block per batch)
// ---------------------------------------------------------------------------
__global__ void k_attention(const float* __restrict__ att1, const float* __restrict__ att2,
                            const float* __restrict__ wfull, const float* __restrict__ bfull,
                            const float* __restrict__ eo, const int* __restrict__ sortind,
                            const int* __restrict__ declen, int t,
                            float* __restrict__ alphas_out, float* __restrict__ awe) {
  __shared__ float s_a[PP];
  __shared__ float s_red[2];
  int b = blockIdx.x;
  int tid = threadIdx.x;
  const float* a2 = att2 + (size_t)b * AD_;
  if (tid < PP) {
    const float* a1 = att1 + ((size_t)b * PP + tid) * AD_;
    float s = 0.0f;
    for (int k = 0; k < AD_; ++k) {
      float u = a1[k] + a2[k];
      u = u > 0.0f ? u : 0.0f;
      s += u * wfull[k];
    }
    s_a[tid] = s + bfull[0];
  }
  __syncthreads();
  if (tid == 0) {
    float mx = s_a[0];
    for (int p = 1; p < PP; ++p) mx = fmaxf(mx, s_a[p]);
    float sum = 0.0f;
    for (int p = 0; p < PP; ++p) sum += __expf(s_a[p] - mx);
    s_red[0] = mx;
    s_red[1] = sum;
  }
  __syncthreads();
  float mask = (declen[b] > t) ? 1.0f : 0.0f;
  if (tid < PP) {
    float al = __expf(s_a[tid] - s_red[0]) / s_red[1];
    s_a[tid] = al;
    alphas_out[((size_t)b * TDEC + t) * PP + tid] = al * mask;
  }
  __syncthreads();
  const float* eorow = eo + (size_t)sortind[b] * PP * ED_;
  float* aw = awe + (size_t)b * ED_;
  for (int d = tid; d < ED_; d += blockDim.x) {
    float s = 0.0f;
    for (int p = 0; p < PP; ++p) s += eorow[(size_t)p * ED_ + d] * s_a[p];
    aw[d] = s;
  }
}

// ---------------------------------------------------------------------------
// 7) x = [emb_t, sigmoid(gatelin)*awe, zero pad to 2368]
// ---------------------------------------------------------------------------
#define XPAD 2368
__global__ void k_assemble_x(const float* __restrict__ embs, int t,
                             const float* __restrict__ gatelin,
                             const float* __restrict__ awe, float* __restrict__ x) {
  int idx = blockIdx.x * blockDim.x + threadIdx.x;
  if (idx >= BB * XPAD) return;
  int b = idx / XPAD;
  int j = idx - b * XPAD;
  float v;
  if (j < EE) {
    v = embs[((size_t)b * TT + t) * EE + j];
  } else if (j < EE + ED_) {
    int d = j - EE;
    float g = sigmoidf_(gatelin[(size_t)b * ED_ + d]);
    v = g * awe[(size_t)b * ED_ + d];
  } else {
    v = 0.0f;
  }
  x[idx] = v;
}

// ---------------------------------------------------------------------------
// 8) LSTM pointwise + length mask (updates h, c in place)
// ---------------------------------------------------------------------------
__global__ void k_lstm(const float* __restrict__ g, float* __restrict__ h,
                       float* __restrict__ c, const int* __restrict__ declen, int t) {
  int idx = blockIdx.x * blockDim.x + threadIdx.x;
  if (idx >= BB * DD_) return;
  int b = idx >> 9;  // /512
  int d = idx & (DD_ - 1);
  const float* gr = g + (size_t)b * 2048;
  float i_ = sigmoidf_(gr[d]);
  float f_ = sigmoidf_(gr[DD_ + d]);
  float gg = tanhf(gr[2 * DD_ + d]);
  float o_ = sigmoidf_(gr[3 * DD_ + d]);
  float cn = f_ * c[idx] + i_ * gg;
  float hn = o_ * tanhf(cn);
  float m = (declen[b] > t) ? 1.0f : 0.0f;
  h[idx] = m * hn + (1.0f - m) * h[idx];
  c[idx] = m * cn + (1.0f - m) * c[idx];
}

// ---------------------------------------------------------------------------
// Host side
// ---------------------------------------------------------------------------
static inline size_t alignUp256(size_t x) { return (x + 255) & ~(size_t)255; }

extern "C" void kernel_launch(void* const* d_in, const int* in_sizes, int n_in,
                              void* d_out, int out_size, void* d_ws, size_t ws_size,
                              hipStream_t stream) {
  (void)in_sizes; (void)n_in; (void)out_size; (void)ws_size;

  const float* encoder_out = (const float*)d_in[0];
  const int*   enc_caps    = (const int*)d_in[1];
  const int*   cap_lens    = (const int*)d_in[2];
  const float* emb         = (const float*)d_in[3];
  const float* W_enc_att   = (const float*)d_in[4];
  const float* b_enc_att   = (const float*)d_in[5];
  const float* W_dec_att   = (const float*)d_in[6];
  const float* b_dec_att   = (const float*)d_in[7];
  const float* w_full      = (const float*)d_in[8];
  const float* b_full      = (const float*)d_in[9];
  const float* W_init_h    = (const float*)d_in[10];
  const float* b_init_h    = (const float*)d_in[11];
  const float* W_init_c    = (const float*)d_in[12];
  const float* b_init_c    = (const float*)d_in[13];
  const float* W_fbeta     = (const float*)d_in[14];
  const float* b_fbeta     = (const float*)d_in[15];
  const float* W_ih        = (const float*)d_in[16];
  const float* b_ih        = (const float*)d_in[17];
  const float* W_hh        = (const float*)d_in[18];
  const float* b_hh        = (const float*)d_in[19];
  const float* W_fc        = (const float*)d_in[20];
  const float* b_fc        = (const float*)d_in[21];

  float* out = (float*)d_out;
  float* out_preds   = out;                                          // 64*95*1000
  float* out_caps    = out + (size_t)BB * TDEC * VV;                 // 64*96
  float* out_declen  = out_caps + (size_t)BB * TT;                   // 64
  float* out_alphas  = out_declen + BB;                              // 64*95*196
  float* out_sortind = out_alphas + (size_t)BB * TDEC * PP;          // 64

  // -------- workspace layout --------
  char* base = (char*)d_ws;
  size_t off = 0;
  auto alloc = [&](size_t bytes) { size_t o = off; off = alignUp256(off + bytes); return o; };

  // swizzled bf16 weights: Kt * ntiles * 512 halfs each
  const int KtENC = 64, NtENC = 32;   // 2048 x 512
  const int KtDA  = 16, NtDA  = 32;   // 512 x 512
  const int KtFB  = 16, NtFB  = 128;  // 512 x 2048
  const int KtIH  = 74, NtIH  = 128;  // 2348(->2368) x 2048
  const int KtHH  = 16, NtHH  = 128;  // 512 x 2048
  const int KtFC  = 16, NtFC  = 63;   // 512 x 1000(->1008)
  const int KtIN  = 64, NtIN  = 32;   // 2048 x 512

  __bf16* sw_enc = (__bf16*)(base + alloc((size_t)KtENC * NtENC * 512 * 2));
  __bf16* sw_da  = (__bf16*)(base + alloc((size_t)KtDA  * NtDA  * 512 * 2));
  __bf16* sw_fb  = (__bf16*)(base + alloc((size_t)KtFB  * NtFB  * 512 * 2));
  __bf16* sw_ih  = (__bf16*)(base + alloc((size_t)KtIH  * NtIH  * 512 * 2));
  __bf16* sw_hh  = (__bf16*)(base + alloc((size_t)KtHH  * NtHH  * 512 * 2));
  __bf16* sw_fc  = (__bf16*)(base + alloc((size_t)KtFC  * NtFC  * 512 * 2));
  __bf16* sw_ih0 = (__bf16*)(base + alloc((size_t)KtIN  * NtIN  * 512 * 2));
  __bf16* sw_ic0 = (__bf16*)(base + alloc((size_t)KtIN  * NtIN  * 512 * 2));

  float* att1    = (float*)(base + alloc((size_t)BB * PP * AD_ * 4));
  float* embs    = (float*)(base + alloc((size_t)BB * TT * EE * 4));
  float* mean_eo = (float*)(base + alloc((size_t)BB * ED_ * 4));
  float* h_buf   = (float*)(base + alloc((size_t)BB * DD_ * 4));
  float* c_buf   = (float*)(base + alloc((size_t)BB * DD_ * 4));
  float* att2    = (float*)(base + alloc((size_t)BB * AD_ * 4));
  float* gatelin = (float*)(base + alloc((size_t)BB * ED_ * 4));
  float* awe     = (float*)(base + alloc((size_t)BB * ED_ * 4));
  float* hg      = (float*)(base + alloc((size_t)BB * 2048 * 4));
  float* gbuf    = (float*)(base + alloc((size_t)BB * 2048 * 4));
  float* x_buf   = (float*)(base + alloc((size_t)BB * XPAD * 4));
  int*   sortind = (int*)(base + alloc((size_t)BB * 4));
  int*   declen  = (int*)(base + alloc((size_t)BB * 4));
  int*   caps_s  = (int*)(base + alloc((size_t)BB * TT * 4));

  auto gemm = [&](const float* A, int lda, const int* si, int gp,
                  const __bf16* W, const float* bias, const float* add, int ldadd,
                  float* C, int ldc, int M, int Kt, int ntiles, int Nout,
                  const int* dl, int t) {
    int mtiles = M / 16;
    int ntg = (ntiles + 3) / 4;
    int waves = mtiles * ntg;
    int blocks = (waves * 32 + 255) / 256;
    k_gemm<<<blocks, 256, 0, stream>>>(A, lda, si, gp, W, bias, add, ldadd, C, ldc,
                                       M, Kt, ntiles, Nout, dl, t);
  };

  // ---- setup ----
  k_sort<<<1, 256, 0, stream>>>(cap_lens, enc_caps, sortind, declen, caps_s,
                                out_caps, out_declen, out_sortind);

  auto conv = [&](const float* src, __bf16* dst, int K, int N, int Kt, int nt) {
    int total = Kt * nt * 512;
    int blocks = (total + 255) / 256;
    k_convert<<<blocks, 256, 0, stream>>>(src, dst, K, N, Kt, nt);
  };
  conv(W_enc_att, sw_enc, 2048, 512,  KtENC, NtENC);
  conv(W_dec_att, sw_da,  512,  512,  KtDA,  NtDA);
  conv(W_fbeta,   sw_fb,  512,  2048, KtFB,  NtFB);
  conv(W_ih,      sw_ih,  2348, 2048, KtIH,  NtIH);
  conv(W_hh,      sw_hh,  512,  2048, KtHH,  NtHH);
  conv(W_fc,      sw_fc,  512,  1000, KtFC,  NtFC);
  conv(W_init_h,  sw_ih0, 2048, 512,  KtIN,  NtIN);
  conv(W_init_c,  sw_ic0, 2048, 512,  KtIN,  NtIN);

  k_embs<<<(BB * TT * EE + 255) / 256, 256, 0, stream>>>(emb, caps_s, embs);
  k_mean<<<(BB * ED_ + 255) / 256, 256, 0, stream>>>(encoder_out, sortind, mean_eo);

  // h0, c0
  gemm(mean_eo, 2048, nullptr, 0, sw_ih0, b_init_h, nullptr, 0,
       h_buf, DD_, BB, KtIN, NtIN, DD_, nullptr, 0);
  gemm(mean_eo, 2048, nullptr, 0, sw_ic0, b_init_c, nullptr, 0,
       c_buf, DD_, BB, KtIN, NtIN, DD_, nullptr, 0);

  // att1 = eo_sorted @ W_enc_att + b  (row indirection through sortind)
  gemm(encoder_out, ED_, sortind, PP, sw_enc, b_enc_att, nullptr, 0,
       att1, AD_, BB * PP, KtENC, NtENC, AD_, nullptr, 0);

  // ---- sequential decode ----
  for (int t = 0; t < TDEC; ++t) {
    // att2 = h @ W_dec_att + b
    gemm(h_buf, DD_, nullptr, 0, sw_da, b_dec_att, nullptr, 0,
         att2, AD_, BB, KtDA, NtDA, AD_, nullptr, 0);
    // gate logits = h @ W_fbeta + b
    gemm(h_buf, DD_, nullptr, 0, sw_fb, b_fbeta, nullptr, 0,
         gatelin, ED_, BB, KtFB, NtFB, ED_, nullptr, 0);
    // hg = h @ W_hh + b_hh
    gemm(h_buf, DD_, nullptr, 0, sw_hh, b_hh, nullptr, 0,
         hg, 2048, BB, KtHH, NtHH, 2048, nullptr, 0);

    k_attention<<<BB, 256, 0, stream>>>(att1, att2, w_full, b_full, encoder_out,
                                        sortind, declen, t, out_alphas, awe);
    k_assemble_x<<<(BB * XPAD + 255) / 256, 256, 0, stream>>>(embs, t, gatelin, awe, x_buf);

    // gates = x @ W_ih + b_ih + hg
    gemm(x_buf, XPAD, nullptr, 0, sw_ih, b_ih, hg, 2048,
         gbuf, 2048, BB, KtIH, NtIH, 2048, nullptr, 0);

    k_lstm<<<(BB * DD_ + 255) / 256, 256, 0, stream>>>(gbuf, h_buf, c_buf, declen, t);

    // preds[:, t, :] = mask * (h @ W_fc + b_fc), written directly into d_out
    gemm(h_buf, DD_, nullptr, 0, sw_fc, b_fc, nullptr, 0,
         out_preds + (size_t)t * VV, TDEC * VV, BB, KtFC, NtFC, VV, declen, t);
  }
}
